// MMFConv2d_7791070675565
// MI455X (gfx1250) — 8-bit matrix operand order repaired and checked by reading, compiled, not run on hardware
//
#include <hip/hip_runtime.h>
#include <hip/hip_bf16.h>

typedef __attribute__((ext_vector_type(8))) int   v8i;
typedef __attribute__((ext_vector_type(8))) float v8f;

#define Bn   8
#define Cch  64
#define Hh   256
#define Ww   256
#define NPIX (Bn * Hh * Ww)

__global__ __launch_bounds__(512, 1)
void quant_w_kernel(const float* __restrict__ W, signed char* __restrict__ wpk,
                    float* __restrict__ winv)
{
    __shared__ float red[512];
    __shared__ float s_w_sh;
    const int tid = threadIdx.x;

    float s = 0.0f;
    for (int i = tid; i < Cch * Cch * 9; i += 512) s += fabsf(W[i]);
    red[tid] = s;
    __syncthreads();
    for (int off = 256; off > 0; off >>= 1) {
        if (tid < off) red[tid] += red[tid + off];
        __syncthreads();
    }
    if (tid == 0) {
        float m = fmaxf(red[0] * (1.0f / (Cch * Cch * 9)), 1e-8f);
        *winv  = m;
        s_w_sh = 1.0f / m;
    }
    __syncthreads();
    const float sw = s_w_sh;

    for (int idx = tid; idx < 9 * 4 * 32 * 32; idx += 512) {
        const int j    = idx & 31;
        const int lane = (idx >> 5) & 31;
        const int g    = idx >> 10;
        const int mt   = g & 3;
        const int t    = g >> 2;
        const int v    = j >> 2, bb = j & 3;
        const int K = ((j >> 4) << 5) + (j & 15) + ((lane & 16) ? 16 : 0);
        const int M = (lane & 15) + (mt << 4);
        const int ky = t / 3, kx = t - ky * 3;
        const float wv = W[(((M << 6) + K) * 3 + ky) * 3 + kx];
        const float qf = fminf(fmaxf(rintf(sw * wv), -1.0f), 1.0f);
        wpk[idx] = (signed char)(int)qf;
    }
}

__global__ __launch_bounds__(256, 4)
void quant_act_kernel(const float* __restrict__ x,
                      signed char* __restrict__ q, float* __restrict__ dq)
{
    const int p  = blockIdx.x * 256 + threadIdx.x;
    const int b  = p >> 16;
    const int hw = p & 65535;
    const float* xp = x + ((size_t)b << 22) + hw;

    float s = 0.0f, ss = 0.0f, mx = -3.402823466e38f, mn = 3.402823466e38f;
    #pragma unroll 8
    for (int c = 0; c < Cch; ++c) {
        const float v = xp[(size_t)c << 16];
        s += v;
        ss = fmaf(v, v, ss);
        mx = fmaxf(mx, v);
        mn = fminf(mn, v);
    }
    const float mu  = s * (1.0f / 64.0f);
    const float var = fmaf(ss, 1.0f / 64.0f, -mu * mu);
    const float r   = rsqrtf(var + 1e-8f);
    const float am  = fmaxf(fmaxf(mx - mu, mu - mn) * r, 1e-8f);
    const float sact = 127.0f / am;
    dq[p] = am * (1.0f / 127.0f);
    const float k = sact * r;

    int4 pk[4];
    int* pw = (int*)pk;
    #pragma unroll 8
    for (int c = 0; c < Cch; ++c) {
        const float v  = xp[(size_t)c << 16];
        float qf = rintf(k * (v - mu));
        qf = fminf(fmaxf(qf, -128.0f), 127.0f);
        const int qi = (int)qf & 255;
        if ((c & 3) == 0) pw[c >> 2]  = qi;
        else              pw[c >> 2] |= qi << ((c & 3) * 8);
    }
    int4* qp = (int4*)(q + ((size_t)p << 6));
    qp[0] = pk[0]; qp[1] = pk[1]; qp[2] = pk[2]; qp[3] = pk[3];
}

__global__ __launch_bounds__(256, 4)
void conv_iu8_kernel(const signed char* __restrict__ q,
                     const float* __restrict__ dq,
                     const signed char* __restrict__ wpk,
                     const float* __restrict__ winv_p,
                     const float* __restrict__ bias,
                     float* __restrict__ out)
{
    const int tid   = threadIdx.x;
    const int lane  = tid & 31;
    const int wave  = tid >> 5;
    const int mt    = wave >> 1;
    const int nsub  = wave & 1;
    const int bi    = blockIdx.x;
    const int row   = bi >> 3;
    const int b     = row >> 8;
    const int h     = row & 255;
    const int w0    = ((bi & 7) << 5) + (nsub << 4);
    const int n     = lane & 15;
    const int wp    = w0 + n;
    const int khalf = (lane >> 4) & 1;

    const float winv = *winv_p;

    {
        const int hn = (h + 1 < 256) ? (h + 1) : 255;
        __builtin_prefetch(q + ((size_t)(((b << 8) + hn) * 256 + w0) << 6), 0, 0);
    }

    v8i amArr[9];
    #pragma unroll
    for (int t = 0; t < 9; ++t) {
        const signed char* ap = wpk + (size_t)((((t << 2) + mt) << 5) + lane) * 32;
        const int4 a0 = *(const int4*)ap;
        const int4 a1 = *(const int4*)(ap + 16);
        v8i am;
        am[0] = a0.x; am[1] = a0.y; am[2] = a0.z; am[3] = a0.w;
        am[4] = a1.x; am[5] = a1.y; am[6] = a1.z; am[7] = a1.w;
        amArr[t] = am;
    }

    v8f acc = {};
    #pragma unroll
    for (int ky = 0; ky < 3; ++ky) {
        const int hh = h + ky - 1;
        const int hc = min(max(hh, 0), 255);
        const float okh = (hh >= 0 && hh < 256) ? 1.0f : 0.0f;
        #pragma unroll
        for (int kx = 0; kx < 3; ++kx) {
            const int t  = ky * 3 + kx;
            const int ww = wp + kx - 1;
            const int wc = min(max(ww, 0), 255);
            const int pin = (((b << 8) + hc) << 8) + wc;
            const float okf = (ww >= 0 && ww < 256) ? okh : 0.0f;
            const float d = dq[pin] * okf;

            const signed char* bp = q + ((size_t)pin << 6) + (khalf << 4);
            const int4 b0 = *(const int4*)bp;
            const int4 b1 = *(const int4*)(bp + 32);
            v8i bm;
            bm[0] = b0.x; bm[1] = b0.y; bm[2] = b0.z; bm[3] = b0.w;
            bm[4] = b1.x; bm[5] = b1.y; bm[6] = b1.z; bm[7] = b1.w;

            v8i z = {};
            v8i rr = __builtin_amdgcn_wmma_i32_16x16x64_iu8(
                true, amArr[t], true, bm, z, false, false);
            #pragma unroll
            for (int i = 0; i < 8; ++i) acc[i] = fmaf((float)rr[i], d, acc[i]);
        }
    }

    const int obase = (mt << 4) + (khalf << 3);
    const float4 bb0 = *(const float4*)(bias + obase);
    const float4 bb1 = *(const float4*)(bias + obase + 4);
    const float bl[8] = { bb0.x, bb0.y, bb0.z, bb0.w, bb1.x, bb1.y, bb1.z, bb1.w };
    float* op = out + (((size_t)b * 64 + obase) << 16) + (h << 8) + wp;
    #pragma unroll
    for (int r8 = 0; r8 < 8; ++r8) {
        op[(size_t)r8 << 16] = fmaf(acc[r8], winv, bl[r8]);
    }
}

extern "C" void kernel_launch(void* const* d_in, const int* in_sizes, int n_in,
                              void* d_out, int out_size, void* d_ws, size_t ws_size,
                              hipStream_t stream)
{
    const float* x    = (const float*)d_in[0];
    const float* W    = (const float*)d_in[1];
    const float* bias = (const float*)d_in[2];

    char* ws = (char*)d_ws;
    signed char* q    = (signed char*)ws;
    float*       dq   = (float*)(ws + 33554432);
    signed char* wpk  = (signed char*)(ws + 35651584);
    float*       winv = (float*)(ws + 35688448);

    quant_w_kernel  <<<1,     512, 0, stream>>>(W, wpk, winv);
    quant_act_kernel<<<NPIX / 256, 256, 0, stream>>>(x, q, dq);
    conv_iu8_kernel <<<NPIX / 32,  256, 0, stream>>>(q, dq, wpk, winv, bias,
                                                     (float*)d_out);
}
